// my_model5_65764539236574
// MI455X (gfx1250) — compile-verified, matrix operand order checked by reading, not run on hardware
//
#include <hip/hip_runtime.h>
#include <math.h>

// ---- problem constants ----
#define B_  4
#define L_  1024
#define D_  768
#define H_  12
#define M_  96
#define E_  48
#define P_  1024
#define BS_ 64
#define NB_ 12

typedef __attribute__((ext_vector_type(16))) _Float16 v16h;
typedef __attribute__((ext_vector_type(8)))  float    v8f;

#define WMMA_F16(a, b, c) \
    __builtin_amdgcn_wmma_f32_16x16x32_f16(false, (a), false, (b), (short)0, (c), false, false)

// ---------------------------------------------------------------------------
// WMMA f16 16x16x32 operand packing (per CDNA5 ISA 7.12.2):
//   A (16xK): lane holds row m = lane&15; lanes 0-15 carry K in {0..7,16..23},
//             lanes 16-31 carry K in {8..15,24..31} of each 32-wide k-tile.
//   B (Kx16): same K pattern per lane, column n = lane&15.
// Packed tile = 512 f16 (1024B); lane's 16 halves are one contiguous 32B load.
// ---------------------------------------------------------------------------
__device__ __forceinline__ int kmap(int e, int lgrp) {
    return e + 8 * lgrp + ((e >= 8) ? 8 : 0);
}

__device__ __forceinline__ void packA_store(_Float16* base, int Ktiles,
                                            int row, int k, float v) {
    int mtile = row >> 4, m = row & 15;
    int ktile = k >> 5,  kk = k & 31;
    int q = kk >> 3;
    int lgrp = q & 1;
    int e = ((q >> 1) << 3) | (kk & 7);
    base[((size_t)(mtile * Ktiles + ktile) << 9) + ((lgrp << 4) + m) * 16 + e] =
        (_Float16)v;
}

// ---------------------------------------------------------------------------
// K1: mention[b,m,d] = exp( sum_l mm[b,m,l] * ctx[b,l,d] )
// grid (12 dtiles, 6 mtiles, B), 256 thr; mm tile staged in LDS (64KB).
// ---------------------------------------------------------------------------
__global__ __launch_bounds__(256)
void mention_kernel(const float* __restrict__ ctx, const float* __restrict__ mm,
                    float* __restrict__ mention) {
    __shared__ float smm[16][L_];
    int b = blockIdx.z, mt = blockIdx.y, dt = blockIdx.x;
    const float* mmb = mm + ((size_t)b * M_ + mt * 16) * L_;
    for (int i = threadIdx.x; i < 16 * L_; i += 256)
        smm[i >> 10][i & 1023] = mmb[i];
    __syncthreads();
    int d  = dt * 64 + (threadIdx.x & 63);
    int m0 = threadIdx.x >> 6;                 // 0..3
    float acc[4] = {0.f, 0.f, 0.f, 0.f};
    const float* cb = ctx + (size_t)b * L_ * D_ + d;
    for (int l = 0; l < L_; ++l) {
        float c = cb[(size_t)l * D_];
#pragma unroll
        for (int i = 0; i < 4; ++i) acc[i] += smm[m0 + 4 * i][l] * c;
    }
    float* ob = mention + ((size_t)b * M_ + mt * 16) * D_ + d;
#pragma unroll
    for (int i = 0; i < 4; ++i) ob[(size_t)(m0 + 4 * i) * D_] = __expf(acc[i]);
}

// ---------------------------------------------------------------------------
// K2: entity[b,e,d] = (row empty) ? 0 : log( sum_m em[b,e,m]*mention[b,m,d] )
// ---------------------------------------------------------------------------
__global__ __launch_bounds__(256)
void entity_kernel(const float* __restrict__ em, const float* __restrict__ mention,
                   float* __restrict__ entity) {
    int be = blockIdx.x, b = be / E_;
    const float* emr = em + (size_t)be * M_;
    float esum = 0.f;
    for (int m = 0; m < M_; ++m) esum += emr[m];
    const float* mb = mention + (size_t)b * M_ * D_;
    for (int j = 0; j < 3; ++j) {
        int d = threadIdx.x + j * 256;
        float s = 0.f;
        for (int m = 0; m < M_; ++m) s += emr[m] * mb[(size_t)m * D_ + d];
        entity[(size_t)be * D_ + d] = (esum == 0.f) ? 0.f : logf(s);
    }
}

// ---------------------------------------------------------------------------
// K3: em_norm row -> packed WMMA-A tiles (E x L per batch, f16)
// ---------------------------------------------------------------------------
__global__ __launch_bounds__(256)
void emnorm_kernel(const float* __restrict__ em, const float* __restrict__ mm,
                   _Float16* __restrict__ emA) {
    __shared__ float red[256];
    int be = blockIdx.x, b = be / E_, e = be % E_;
    const float* emr = em + (size_t)be * M_;
    const float* mmb = mm + (size_t)b * M_ * L_;
    float vals[4]; float lsum = 0.f;
    for (int j = 0; j < 4; ++j) {
        int l = threadIdx.x + j * 256;
        float s = 0.f;
        for (int m = 0; m < M_; ++m) s += emr[m] * mmb[(size_t)m * L_ + l];
        vals[j] = s; lsum += s;
    }
    red[threadIdx.x] = lsum; __syncthreads();
    for (int off = 128; off > 0; off >>= 1) {
        if (threadIdx.x < off) red[threadIdx.x] += red[threadIdx.x + off];
        __syncthreads();
    }
    float scale = 1.f / (red[0] + 1e-20f);
    _Float16* base = emA + (size_t)b * 96 * 512;   // 3 mtiles * 32 ktiles
    for (int j = 0; j < 4; ++j) {
        int l = threadIdx.x + j * 256;
        packA_store(base, 32, e, l, vals[j] * scale);
    }
}

// ---------------------------------------------------------------------------
// K4: generic strided fp32 -> packed WMMA-B f16 tiles.
//     logical B[k][n] = src[k*sk + n*sn]; tile id = ktile*Nt + ntile.
// ---------------------------------------------------------------------------
__global__ __launch_bounds__(256)
void packB_kernel(const float* __restrict__ src, _Float16* __restrict__ dst,
                  int Nt, int sk, int sn,
                  size_t srcBatchStride, size_t dstBatchStride) {
    int tile = blockIdx.x, bat = blockIdx.y;
    int ktile = tile / Nt, ntile = tile % Nt;
    const float* s = src + srcBatchStride * bat;
    _Float16*   d = dst + dstBatchStride * bat + ((size_t)tile << 9);
    for (int idx = threadIdx.x; idx < 512; idx += 256) {
        int lane = idx >> 4, e = idx & 15;
        int n15 = lane & 15, lgrp = lane >> 4;
        int k = ktile * 32 + kmap(e, lgrp);
        int n = ntile * 16 + n15;
        d[idx] = (_Float16)s[(size_t)k * sk + (size_t)n * sn];
    }
}

// ---------------------------------------------------------------------------
// K5: gather h/t rows from entity (masked) straight into packed A layout
// ---------------------------------------------------------------------------
__global__ __launch_bounds__(256)
void gather_ht_kernel(const float* __restrict__ entity, const int* __restrict__ hts,
                      _Float16* __restrict__ hA, _Float16* __restrict__ tA) {
    int bp = blockIdx.x, b = bp >> 10, p = bp & 1023;
    int hi = hts[(size_t)bp * 2], ti = hts[(size_t)bp * 2 + 1];
    float mask = (hi + ti != 0) ? 1.f : 0.f;
    const float* eh = entity + ((size_t)b * E_ + hi) * D_;
    const float* et = entity + ((size_t)b * E_ + ti) * D_;
    _Float16* hb = hA + (size_t)b * 64 * 24 * 512;
    _Float16* tb = tA + (size_t)b * 64 * 24 * 512;
    for (int j = 0; j < 3; ++j) {
        int d = threadIdx.x + j * 256;
        packA_store(hb, 24, p, d, eh[d] * mask);
        packA_store(tb, 24, p, d, et[d] * mask);
    }
}

// ---------------------------------------------------------------------------
// K6: ent_att[bh, E, L] = em_norm(ExL) x attention(LxL), WMMA f16->f32.
// Attention read once from HBM, fp32 -> f16 inline; per-lane b32 loads
// coalesce along N. Each B tile amortized over all 3 E-mtiles.
// grid (8, B*H), 8 waves/block: wave -> ntile.
// ---------------------------------------------------------------------------
__global__ __launch_bounds__(256)
void entatt_kernel(const _Float16* __restrict__ emA, const float* __restrict__ att,
                   float* __restrict__ entatt) {
    int bh   = blockIdx.y;
    int b    = bh / H_;
    int wid  = threadIdx.x >> 5;
    int lane = threadIdx.x & 31;
    int ntile = blockIdx.x * 8 + wid;          // 0..63
    int n15 = lane & 15, lgrp = lane >> 4;
    const _Float16* Ab  = emA + (size_t)b * 96 * 512 + lane * 16;
    const float*    atb = att + (size_t)bh * L_ * L_ + ntile * 16 + n15;
    v8f c0 = {}, c1 = {}, c2 = {};
    int krow[16];
#pragma unroll
    for (int e = 0; e < 16; ++e) krow[e] = kmap(e, lgrp);
    for (int kt = 0; kt < 32; ++kt) {
        union { v16h v; _Float16 h[16]; } bu;
#pragma unroll
        for (int e = 0; e < 16; ++e)
            bu.h[e] = (_Float16)atb[(size_t)(kt * 32 + krow[e]) * L_];
        v16h a0 = *(const v16h*)(Ab + (size_t)(0 * 32 + kt) * 512);
        v16h a1 = *(const v16h*)(Ab + (size_t)(1 * 32 + kt) * 512);
        v16h a2 = *(const v16h*)(Ab + (size_t)(2 * 32 + kt) * 512);
        c0 = WMMA_F16(a0, bu.v, c0);
        c1 = WMMA_F16(a1, bu.v, c1);
        c2 = WMMA_F16(a2, bu.v, c2);
    }
    float* ob = entatt + (size_t)bh * E_ * L_ + ntile * 16 + n15;
#pragma unroll
    for (int r = 0; r < 8; ++r) {
        int ro = r + 8 * lgrp;                 // C layout: vgpr r -> rows r / r+8
        ob[(size_t)(ro) * L_]      = c0[r];
        ob[(size_t)(16 + ro) * L_] = c1[r];
        ob[(size_t)(32 + ro) * L_] = c2[r];
    }
}

// ---------------------------------------------------------------------------
// K7: ctx_att[b,p,l] = sum_h ent_att[h,hts0,l]*ent_att[h,hts1,l], normalized,
//     written directly as packed WMMA-A tiles (P x L).
// ---------------------------------------------------------------------------
__global__ __launch_bounds__(256)
void ctxatt_kernel(const float* __restrict__ entatt, const int* __restrict__ hts,
                   _Float16* __restrict__ ctxA) {
    __shared__ float red[256];
    int bp = blockIdx.x, b = bp >> 10, p = bp & 1023;
    int hi = hts[(size_t)bp * 2], ti = hts[(size_t)bp * 2 + 1];
    const float* ea = entatt + (size_t)b * H_ * E_ * L_;
    float vals[4]; float lsum = 0.f;
    for (int j = 0; j < 4; ++j) {
        int l = threadIdx.x + j * 256;
        float s = 0.f;
        for (int h = 0; h < H_; ++h) {
            const float* rh = ea + ((size_t)h * E_ + hi) * L_;
            const float* rt = ea + ((size_t)h * E_ + ti) * L_;
            s += rh[l] * rt[l];
        }
        vals[j] = s; lsum += s;
    }
    red[threadIdx.x] = lsum; __syncthreads();
    for (int off = 128; off > 0; off >>= 1) {
        if (threadIdx.x < off) red[threadIdx.x] += red[threadIdx.x + off];
        __syncthreads();
    }
    float scale = 1.f / (red[0] + 1e-20f);
    _Float16* base = ctxA + (size_t)b * 64 * 32 * 512;
    for (int j = 0; j < 4; ++j) {
        int l = threadIdx.x + j * 256;
        packA_store(base, 32, p, l, vals[j] * scale);
    }
}

// ---------------------------------------------------------------------------
// K8: context_info = ctx_att(PxL) x context(LxD), masked, output packed-A.
// 2x2 register blocking: 4 WMMAs per 4 operand loads -> halves L2 traffic.
// jobs = B * (64/2) * (48/2) = 3072, 8 waves/block -> 384 blocks.
// ---------------------------------------------------------------------------
__global__ __launch_bounds__(256)
void gemm_ci_kernel(const _Float16* __restrict__ ctxA, const _Float16* __restrict__ ctxB,
                    const int* __restrict__ hts, _Float16* __restrict__ ciA) {
    int job  = blockIdx.x * 8 + (threadIdx.x >> 5);
    int lane = threadIdx.x & 31;
    int b = job / (32 * 24), rem = job % (32 * 24);
    int mt2 = rem / 24, nt2 = rem % 24;
    int mtile0 = mt2 * 2, ntile0 = nt2 * 2;
    const _Float16* Abase = ctxA + (size_t)b * 64 * 32 * 512 + lane * 16;
    const _Float16* Bbase = ctxB + (size_t)b * 32 * 48 * 512 + lane * 16;
    v8f c00 = {}, c01 = {}, c10 = {}, c11 = {};
    for (int kt = 0; kt < 32; ++kt) {
        v16h a0 = *(const v16h*)(Abase + (size_t)(mtile0 * 32 + kt) * 512);
        v16h a1 = *(const v16h*)(Abase + (size_t)((mtile0 + 1) * 32 + kt) * 512);
        v16h b0 = *(const v16h*)(Bbase + (size_t)(kt * 48 + ntile0) * 512);
        v16h b1 = *(const v16h*)(Bbase + (size_t)(kt * 48 + ntile0 + 1) * 512);
        c00 = WMMA_F16(a0, b0, c00);
        c01 = WMMA_F16(a0, b1, c01);
        c10 = WMMA_F16(a1, b0, c10);
        c11 = WMMA_F16(a1, b1, c11);
    }
    int n15 = lane & 15, lgrp = lane >> 4;
    _Float16* cb = ciA + (size_t)b * 64 * 24 * 512;
#pragma unroll
    for (int mi = 0; mi < 2; ++mi) {
#pragma unroll
        for (int r = 0; r < 8; ++r) {
            int row = (mtile0 + mi) * 16 + r + 8 * lgrp;
            int hi = hts[((size_t)b * P_ + row) * 2];
            int ti = hts[((size_t)b * P_ + row) * 2 + 1];
            float mask = (hi + ti != 0) ? 1.f : 0.f;
            float v0 = (mi == 0) ? c00[r] : c10[r];
            float v1 = (mi == 0) ? c01[r] : c11[r];
            packA_store(cb, 24, row, ntile0 * 16 + n15,      v0 * mask);
            packA_store(cb, 24, row, (ntile0 + 1) * 16 + n15, v1 * mask);
        }
    }
}

// ---------------------------------------------------------------------------
// K9: x2 = tanh( x*W^T + ci*Wc^T + bias1 + bias2 )  (dual-WMMA, 2x2 blocked)
// jobs = B * 32 * 24 = 3072 -> 384 blocks; 8 WMMAs per 8 loads per k-step.
// ---------------------------------------------------------------------------
__global__ __launch_bounds__(256)
void gemm_proj_kernel(const _Float16* __restrict__ xA, const _Float16* __restrict__ ciA,
                      const _Float16* __restrict__ WB, const _Float16* __restrict__ WcB,
                      const float* __restrict__ bias1, const float* __restrict__ bias2,
                      float* __restrict__ out) {
    int job  = blockIdx.x * 8 + (threadIdx.x >> 5);
    int lane = threadIdx.x & 31;
    int b = job / (32 * 24), rem = job % (32 * 24);
    int mt2 = rem / 24, nt2 = rem % 24;
    int mtile0 = mt2 * 2, ntile0 = nt2 * 2;
    const _Float16* A1 = xA  + (size_t)b * 64 * 24 * 512 + lane * 16;
    const _Float16* A2 = ciA + (size_t)b * 64 * 24 * 512 + lane * 16;
    const _Float16* B1 = WB  + lane * 16;
    const _Float16* B2 = WcB + lane * 16;
    v8f c00 = {}, c01 = {}, c10 = {}, c11 = {};
    for (int kt = 0; kt < 24; ++kt) {
        size_t am0 = (size_t)(mtile0 * 24 + kt) * 512;
        size_t am1 = (size_t)((mtile0 + 1) * 24 + kt) * 512;
        size_t bn0 = (size_t)(kt * 48 + ntile0) * 512;
        size_t bn1 = (size_t)(kt * 48 + ntile0 + 1) * 512;
        v16h x0 = *(const v16h*)(A1 + am0);
        v16h x1 = *(const v16h*)(A1 + am1);
        v16h w0 = *(const v16h*)(B1 + bn0);
        v16h w1 = *(const v16h*)(B1 + bn1);
        c00 = WMMA_F16(x0, w0, c00);
        c01 = WMMA_F16(x0, w1, c01);
        c10 = WMMA_F16(x1, w0, c10);
        c11 = WMMA_F16(x1, w1, c11);
        v16h y0 = *(const v16h*)(A2 + am0);
        v16h y1 = *(const v16h*)(A2 + am1);
        v16h u0 = *(const v16h*)(B2 + bn0);
        v16h u1 = *(const v16h*)(B2 + bn1);
        c00 = WMMA_F16(y0, u0, c00);
        c01 = WMMA_F16(y0, u1, c01);
        c10 = WMMA_F16(y1, u0, c10);
        c11 = WMMA_F16(y1, u1, c11);
    }
    int n15 = lane & 15, lgrp = lane >> 4;
    int col0 = ntile0 * 16 + n15;
    int col1 = (ntile0 + 1) * 16 + n15;
    float bs0 = bias1[col0] + bias2[col0];
    float bs1 = bias1[col1] + bias2[col1];
#pragma unroll
    for (int r = 0; r < 8; ++r) {
        int row0 = mtile0 * 16 + r + 8 * lgrp;
        int row1 = (mtile0 + 1) * 16 + r + 8 * lgrp;
        out[((size_t)b * P_ + row0) * D_ + col0] = tanhf(c00[r] + bs0);
        out[((size_t)b * P_ + row0) * D_ + col1] = tanhf(c01[r] + bs1);
        out[((size_t)b * P_ + row1) * D_ + col0] = tanhf(c10[r] + bs0);
        out[((size_t)b * P_ + row1) * D_ + col1] = tanhf(c11[r] + bs1);
    }
}

// ---------------------------------------------------------------------------
// K10: bin_res[b,p,o] = sum_n b1[n,:] Wb[o,n] b2[n,:] + bbin[o]
// 8 pairs/block to amortize Wb reads; deterministic LDS tree reduction.
// ---------------------------------------------------------------------------
__global__ __launch_bounds__(256)
void bilinear_kernel(const float* __restrict__ h2, const float* __restrict__ t2,
                     const float* __restrict__ Wbin, const float* __restrict__ bbin,
                     float* __restrict__ out) {
    __shared__ float sh[8][D_];
    __shared__ float st[8][D_];
    __shared__ float sred[16][256];
    int gp0 = blockIdx.x * 8;
    for (int i = threadIdx.x; i < 8 * D_; i += 256) {
        int j8 = i / D_, d = i % D_;
        sh[j8][d] = h2[((size_t)gp0 + j8) * D_ + d];
        st[j8][d] = t2[((size_t)gp0 + j8) * D_ + d];
    }
    __syncthreads();
    float local[16];
#pragma unroll
    for (int i = 0; i < 16; ++i) local[i] = 0.f;
    for (int rid = threadIdx.x; rid < 2 * NB_ * BS_; rid += 256) {   // 6 iters
        int o = rid / (NB_ * BS_);
        int rem = rid % (NB_ * BS_);
        int n = rem >> 6, i = rem & 63;
        const float* wr = Wbin + (size_t)o * D_ * BS_ + (size_t)n * BS_ * BS_ + (size_t)i * BS_;
        float w[64];
#pragma unroll
        for (int j = 0; j < 64; ++j) w[j] = wr[j];
#pragma unroll
        for (int j8 = 0; j8 < 8; ++j8) {
            float s = 0.f;
            const float* tr = &st[j8][n * 64];
#pragma unroll
            for (int j = 0; j < 64; ++j) s += w[j] * tr[j];
            local[j8 * 2 + o] += sh[j8][n * 64 + i] * s;
        }
    }
#pragma unroll
    for (int i = 0; i < 16; ++i) sred[i][threadIdx.x] = local[i];
    __syncthreads();
    for (int off = 128; off > 0; off >>= 1) {
        if (threadIdx.x < off)
#pragma unroll
            for (int i = 0; i < 16; ++i)
                sred[i][threadIdx.x] += sred[i][threadIdx.x + off];
        __syncthreads();
    }
    if (threadIdx.x < 16) {
        int j8 = threadIdx.x >> 1, o = threadIdx.x & 1;
        out[((size_t)gp0 + j8) * 2 + o] = sred[threadIdx.x][0] + bbin[o];
    }
}

// ---------------------------------------------------------------------------
extern "C" void kernel_launch(void* const* d_in, const int* in_sizes, int n_in,
                              void* d_out, int out_size, void* d_ws, size_t ws_size,
                              hipStream_t stream) {
    const float* ctx   = (const float*)d_in[0];
    const float* att   = (const float*)d_in[1];
    const float* mm    = (const float*)d_in[2];
    const float* em    = (const float*)d_in[3];
    const int*   hts   = (const int*)  d_in[4];
    const float* Wh    = (const float*)d_in[5];
    const float* bh    = (const float*)d_in[6];
    const float* Whc   = (const float*)d_in[7];
    const float* bhc   = (const float*)d_in[8];
    const float* Wt    = (const float*)d_in[9];
    const float* bt    = (const float*)d_in[10];
    const float* Wtc   = (const float*)d_in[11];
    const float* btc   = (const float*)d_in[12];
    const float* Wbin  = (const float*)d_in[13];
    const float* bbin  = (const float*)d_in[14];
    float* out = (float*)d_out;

    // ---- workspace carve-up (all sizes multiples of 256B) ----
    char* ws = (char*)d_ws;
    size_t off = 0;
    auto alloc = [&](size_t bytes) { char* p = ws + off; off += bytes; return p; };
    float*    mention = (float*)   alloc((size_t)B_ * M_ * D_ * 4);          // 1.13 MB
    float*    entity  = (float*)   alloc((size_t)B_ * E_ * D_ * 4);          // 0.56 MB
    _Float16* emA     = (_Float16*)alloc((size_t)B_ * 96 * 512 * 2);         // 0.38 MB
    _Float16* ctxB    = (_Float16*)alloc((size_t)B_ * 32 * 48 * 512 * 2);    // 6.29 MB
    _Float16* WhB     = (_Float16*)alloc((size_t)24 * 48 * 512 * 2);         // 1.18 MB
    _Float16* WhcB    = (_Float16*)alloc((size_t)24 * 48 * 512 * 2);
    _Float16* WtB     = (_Float16*)alloc((size_t)24 * 48 * 512 * 2);
    _Float16* WtcB    = (_Float16*)alloc((size_t)24 * 48 * 512 * 2);
    float*    entatt  = (float*)   alloc((size_t)B_ * H_ * E_ * L_ * 4);     // 9.44 MB
    _Float16* hA      = (_Float16*)alloc((size_t)B_ * 64 * 24 * 512 * 2);    // 6.29 MB
    _Float16* tA      = (_Float16*)alloc((size_t)B_ * 64 * 24 * 512 * 2);
    _Float16* ctxA    = (_Float16*)alloc((size_t)B_ * 64 * 32 * 512 * 2);    // 8.39 MB
    _Float16* ciA     = (_Float16*)alloc((size_t)B_ * 64 * 24 * 512 * 2);
    float*    h2      = (float*)   alloc((size_t)B_ * P_ * D_ * 4);          // 12.6 MB
    float*    t2      = (float*)   alloc((size_t)B_ * P_ * D_ * 4);
    if (off > ws_size) return;   // workspace too small; bail safely

    // 1-3: small prep (mention / entity / em_norm->packedA)
    mention_kernel<<<dim3(12, 6, B_), 256, 0, stream>>>(ctx, mm, mention);
    entity_kernel<<<B_ * E_, 256, 0, stream>>>(em, mention, entity);
    emnorm_kernel<<<B_ * E_, 256, 0, stream>>>(em, mm, emA);

    // 4: pack B operands (context per batch, 4 weight matrices as W^T)
    packB_kernel<<<dim3(32 * 48, B_), 256, 0, stream>>>(ctx, ctxB, 48, D_, 1,
                                                        (size_t)L_ * D_, (size_t)32 * 48 * 512);
    packB_kernel<<<dim3(24 * 48, 1), 256, 0, stream>>>(Wh,  WhB,  48, 1, D_, 0, 0);
    packB_kernel<<<dim3(24 * 48, 1), 256, 0, stream>>>(Whc, WhcB, 48, 1, D_, 0, 0);
    packB_kernel<<<dim3(24 * 48, 1), 256, 0, stream>>>(Wt,  WtB,  48, 1, D_, 0, 0);
    packB_kernel<<<dim3(24 * 48, 1), 256, 0, stream>>>(Wtc, WtcB, 48, 1, D_, 0, 0);

    // 5: masked h/t gather -> packed A
    gather_ht_kernel<<<B_ * P_, 256, 0, stream>>>(entity, hts, hA, tA);

    // 6: big WMMA GEMM over attention (192MB streamed once from HBM)
    entatt_kernel<<<dim3(8, B_ * H_), 256, 0, stream>>>(emA, att, entatt);

    // 7: pairwise context attention + normalization -> packed A
    ctxatt_kernel<<<B_ * P_, 256, 0, stream>>>(entatt, hts, ctxA);

    // 8: context_info WMMA GEMM (masked), output packed A (2x2 blocked)
    gemm_ci_kernel<<<(B_ * 32 * 24) / 8, 256, 0, stream>>>(ctxA, ctxB, hts, ciA);

    // 9: fused dual-GEMM projections with tanh (2x2 blocked)
    gemm_proj_kernel<<<(B_ * 32 * 24) / 8, 256, 0, stream>>>(hA, ciA, WhB, WhcB, bh, bhc, h2);
    gemm_proj_kernel<<<(B_ * 32 * 24) / 8, 256, 0, stream>>>(tA, ciA, WtB, WtcB, bt, btc, t2);

    // 10: block-diagonal bilinear head
    bilinear_kernel<<<(B_ * P_) / 8, 256, 0, stream>>>(h2, t2, Wbin, bbin, out);
}
